// MambaBlock_81853486727588
// MI455X (gfx1250) — matrix operand order repaired and checked by reading, not compiled, not run on hardware
//
#include <hip/hip_runtime.h>
#include <cstdint>
#include <cstddef>

// Problem constants (from reference):
#define B_      2
#define T_      2048
#define E_      1024
#define INNER_  2048
#define N_      16
#define R_      128
#define F_IN    (2 * INNER_)   // 4096
#define F_X     (R_ + 2 * N_)  // 160
#define BT_     (B_ * T_)      // 4096

typedef __bf16 bf16;
typedef __attribute__((ext_vector_type(16))) __bf16 v16bf;
typedef __attribute__((ext_vector_type(8)))  float  v8f;

__device__ __forceinline__ bf16 f2bf(float f) {
  union { float f; uint32_t u; } in; in.f = f;
  uint32_t r = in.u + 0x7FFFu + ((in.u >> 16) & 1u);   // round-to-nearest-even
  union { uint16_t s; bf16 b; } out; out.s = (uint16_t)(r >> 16);
  return out.b;
}
__device__ __forceinline__ float bf2f(bf16 b) {
  union { uint16_t s; bf16 b; } in; in.b = b;
  union { uint32_t u; float f; } out; out.u = ((uint32_t)in.s) << 16;
  return out.f;
}
__device__ __forceinline__ float sigmoid_(float x) { return 1.f / (1.f + __expf(-x)); }

// CDNA5 async global->LDS copy (ASYNCcnt path). LDS destination offset comes from the VDST VGPR;
// flat shared addresses carry the LDS offset in addr[31:0], so low 32 bits of a
// __shared__ pointer give the VDST value.
__device__ __forceinline__ void async_copy_b128(uint32_t lds_off, const void* gaddr) {
  asm volatile("global_load_async_to_lds_b128 %0, %1, off"
               :: "v"(lds_off), "v"(gaddr) : "memory");
}
__device__ __forceinline__ void wait_async0() {
  asm volatile("s_wait_asynccnt 0x0" ::: "memory");
}
__device__ __forceinline__ uint32_t lds_off32(const void* p) {
  return (uint32_t)(uintptr_t)p;   // addr[31:0] == logical LDS offset
}

// ---------------------------------------------------------------- f32 -> bf16
__global__ __launch_bounds__(256)
void cvt_bf16(const float* __restrict__ in, bf16* __restrict__ out, int n) {
  int i = blockIdx.x * 256 + threadIdx.x;
  if (i < n) out[i] = f2bf(in[i]);
}

// ---------------------------------------------------------------- LayerNorm
__global__ __launch_bounds__(256)
void ln_kernel(const float* __restrict__ x, const float* __restrict__ g,
               const float* __restrict__ be, bf16* __restrict__ out) {
  __shared__ float ss[256], sq[256];
  const int row = blockIdx.x;                 // b*T + t
  const float* xr = x + (size_t)row * E_;
  float s = 0.f, q = 0.f;
  for (int i = threadIdx.x; i < E_; i += 256) { float v = xr[i]; s += v; q += v * v; }
  ss[threadIdx.x] = s; sq[threadIdx.x] = q;
  __syncthreads();
  for (int st = 128; st > 0; st >>= 1) {
    if (threadIdx.x < st) {
      ss[threadIdx.x] += ss[threadIdx.x + st];
      sq[threadIdx.x] += sq[threadIdx.x + st];
    }
    __syncthreads();
  }
  const float mu  = ss[0] * (1.f / E_);
  const float var = sq[0] * (1.f / E_) - mu * mu;
  const float rs  = rsqrtf(var + 1e-5f);
  for (int i = threadIdx.x; i < E_; i += 256)
    out[(size_t)row * E_ + i] = f2bf((xr[i] - mu) * rs * g[i] + be[i]);
}

// ---------------------------------------------------------------- WMMA GEMM
// Y[m, n] = sum_k A[m,k] * W[n,k]   (A: M x K bf16 row-major, W: F x K bf16 row-major)
// Block: 8 waves, tile 128(M) x NT*16(N). B tile (NT*16 x 32K) staged in LDS via
// async copies, double buffered; A fragments prefetched one chunk ahead in regs;
// all NT B fragments ds-loaded before the WMMA burst so waits pipeline.
// Epilogues: 0 = store bf16 (xz)
//            1 = store f32 (dbl) + bf16 of cols < ld_bf (dblR)
//            2 = softplus(acc + aux[col]) -> f32 (dt)
//            3 = acc + aux[row,col] (residual) -> f32 (final out)
enum { EPI_BF16 = 0, EPI_DBL = 1, EPI_DT = 2, EPI_RESID = 3 };

template<int NT, int EPI>
__global__ __launch_bounds__(256)
void gemm_wmma(const bf16* __restrict__ A, const bf16* __restrict__ W, int K,
               float* __restrict__ out_f32, bf16* __restrict__ out_bf16,
               const float* __restrict__ aux, int ld_out, int ld_bf) {
  // B tile: NT*16 rows x 32 K-values x 2B = NT*1024 bytes, double buffered.
  __shared__ __align__(16) uint8_t sB[2][NT * 16 * 64];

  const int tid  = threadIdx.x;
  const int lane = tid & 31;
  const int wave = tid >> 5;
  const int m0   = blockIdx.y * 128 + wave * 16;
  const int n0   = blockIdx.x * (NT * 16);
  const int half = lane >> 4;     // 0: lanes 0-15, 1: lanes 16-31
  const int l16  = lane & 15;

  // Cooperative async-copy mapping: thread copies 16B of the B tile.
  // NT==4 -> all 256 threads copy (predicate folds away, no EXEC juggling).
  const int  crow   = tid >> 2;            // row within tile (needs NT*64 threads)
  const int  celem  = (tid & 3) << 3;      // bf16 element offset inside 32-K row
  const bool copier = (NT >= 4) || (tid < NT * 64);
  const bf16*    gB   = W + (size_t)(n0 + crow) * K + celem;
  const uint32_t ldsB = (crow << 6) + (celem << 1);   // byte offset inside buffer

  // A frag (16x32 bf16): lane holds row M=l16; v0..3 = K 8*half+{0..7}, v4..7 = +16.
  const bf16* arow = A + (size_t)(m0 + l16) * K + half * 8;

  v8f acc[NT] = {};

  // Prologue: stage chunk 0, load A chunk 0.
  if (copier) async_copy_b128(lds_off32(&sB[0][ldsB]), gB);
  uint4 a0 = *(const uint4*)(arow);
  uint4 a1 = *(const uint4*)(arow + 16);
  wait_async0();
  __syncthreads();

  const int NC = K >> 5;
  for (int c = 0; c < NC; ++c) {
    const int cur = c & 1;
    const int k0  = c << 5;
    uint4 a0n, a1n;
    if (c + 1 < NC) {
      if (copier) async_copy_b128(lds_off32(&sB[cur ^ 1][ldsB]), gB + k0 + 32);
      a0n = *(const uint4*)(arow + k0 + 32);
      a1n = *(const uint4*)(arow + k0 + 48);
    }
    union { v16bf v; uint4 u[2]; } a; a.u[0] = a0; a.u[1] = a1;

    // Issue all B-fragment LDS loads first, then the WMMA burst.
    union { v16bf v; uint4 u[2]; } b[NT];
#pragma unroll
    for (int nt = 0; nt < NT; ++nt) {
      // B frag: lane holds col N=l16; v0..3 = K 8*half+{0..7}, v4..7 = +16.
      const uint8_t* bp = &sB[cur][((nt * 16 + l16) << 6) + half * 16];
      b[nt].u[0] = *(const uint4*)(bp);
      b[nt].u[1] = *(const uint4*)(bp + 32);
    }
#pragma unroll
    for (int nt = 0; nt < NT; ++nt)
      acc[nt] = __builtin_amdgcn_wmma_f32_16x16x32_bf16(
          false, a.v, false, b[nt].v, (short)0, acc[nt], false, false);

    if (c + 1 < NC) {
      a0 = a0n; a1 = a1n;
      wait_async0();              // chunk c+1 landed in LDS
    }
    __syncthreads();              // all waves done reading sB[cur] / copy visible
  }

  // C/D layout: VGPR r -> row m0 + 8*half + r, col n0 + nt*16 + l16
#pragma unroll
  for (int nt = 0; nt < NT; ++nt) {
#pragma unroll
    for (int r = 0; r < 8; ++r) {
      const int row = m0 + half * 8 + r;
      const int col = n0 + nt * 16 + l16;
      const float v = acc[nt][r];
      if (EPI == EPI_BF16) {
        out_bf16[(size_t)row * ld_bf + col] = f2bf(v);
      } else if (EPI == EPI_DBL) {
        out_f32[(size_t)row * ld_out + col] = v;
        if (col < ld_bf) out_bf16[(size_t)row * ld_bf + col] = f2bf(v);
      } else if (EPI == EPI_DT) {
        const float s = v + aux[col];                       // + bdt
        out_f32[(size_t)row * ld_out + col] = (s > 20.f) ? s : log1pf(__expf(s));
      } else {                                              // EPI_RESID
        out_f32[(size_t)row * ld_out + col] = v + aux[(size_t)row * ld_out + col];
      }
    }
  }
}

// ------------------------------------------------- depthwise causal conv + SiLU
__global__ __launch_bounds__(256)
void conv_silu_kernel(const bf16* __restrict__ xz, const float* __restrict__ cw,
                      const float* __restrict__ cb, bf16* __restrict__ u) {
  const size_t i = (size_t)blockIdx.x * 256 + threadIdx.x;  // exactly BT_*INNER_ threads
  const int d     = (int)(i & (INNER_ - 1));
  const size_t bt = i >> 11;                                // b*T + t
  const int t     = (int)(bt & (T_ - 1));
  const size_t b0 = bt - t;                                 // b*T
  float acc = cb[d];
  const float* w = cw + d * 4;                              // conv_w (INNER,1,4)
#pragma unroll
  for (int k = 0; k < 4; ++k) {
    const int tt = t - 3 + k;
    if (tt >= 0) acc += w[k] * bf2f(xz[(b0 + tt) * (size_t)F_IN + d]);
  }
  acc = acc * sigmoid_(acc);                                // SiLU
  u[i] = f2bf(acc);
}

// ---------------------------------------------------------------- selective scan
// One lane per (b, d, n): wave32 = 2 channels x 16 states. y reduced over n via
// xor-shuffles within each 16-lane group; fuse + u*Dp and silu(z) gate -> y bf16.
__global__ __launch_bounds__(256)
void scan_kernel(const float* __restrict__ dt, const bf16* __restrict__ u,
                 const float* __restrict__ dbl, const bf16* __restrict__ xz,
                 const float* __restrict__ A_log, const float* __restrict__ Dp,
                 bf16* __restrict__ y) {
  const int tid   = blockIdx.x * 256 + threadIdx.x;
  const int lane  = tid & 31;
  const int wave  = tid >> 5;                 // 0 .. 2047
  const int b     = wave >> 10;               // / (INNER/2)
  const int dpair = wave & 1023;
  const int n     = lane & 15;
  const int d     = (dpair << 1) + (lane >> 4);

  const float A   = -__expf(A_log[d * N_ + n]);
  const float Dpd = Dp[d];
  float h = 0.f;

  for (int t = 0; t < T_; ++t) {
    const size_t base = (size_t)b * T_ + t;
    const float dtv = dt[base * INNER_ + d];
    const float uv  = bf2f(u[base * INNER_ + d]);
    const float Bv  = dbl[base * F_X + R_ + n];
    const float Cv  = dbl[base * F_X + R_ + N_ + n];

    h = __expf(dtv * A) * h + (dtv * uv) * Bv;
    float p = h * Cv;
    p += __shfl_xor(p, 1, 32);
    p += __shfl_xor(p, 2, 32);
    p += __shfl_xor(p, 4, 32);
    p += __shfl_xor(p, 8, 32);                // full sum over 16 n-lanes
    if (n == 0) {
      const float z  = bf2f(xz[base * F_IN + INNER_ + d]);
      const float yv = (p + uv * Dpd) * (z * sigmoid_(z));
      y[base * INNER_ + d] = f2bf(yv);
    }
  }
}

// ---------------------------------------------------------------- launcher
extern "C" void kernel_launch(void* const* d_in, const int* in_sizes, int n_in,
                              void* d_out, int out_size, void* d_ws, size_t ws_size,
                              hipStream_t stream) {
  const float* x      = (const float*)d_in[0];
  const float* norm_g = (const float*)d_in[1];
  const float* norm_b = (const float*)d_in[2];
  const float* W_in   = (const float*)d_in[3];
  const float* conv_w = (const float*)d_in[4];
  const float* conv_b = (const float*)d_in[5];
  const float* Wx     = (const float*)d_in[6];
  const float* Wdt    = (const float*)d_in[7];
  const float* bdt    = (const float*)d_in[8];
  const float* A_log  = (const float*)d_in[9];
  const float* Dp     = (const float*)d_in[10];
  const float* W_out  = (const float*)d_in[11];
  float* out = (float*)d_out;

  char* wp = (char*)d_ws;
  auto alloc = [&](size_t bytes) { char* p = wp; wp += (bytes + 255) & ~(size_t)255; return p; };

  bf16*  xn_b   = (bf16*) alloc((size_t)BT_   * E_     * 2);  //  8.4 MB
  bf16*  Win_b  = (bf16*) alloc((size_t)F_IN  * E_     * 2);  //  8.4 MB
  bf16*  Wx_b   = (bf16*) alloc((size_t)F_X   * INNER_ * 2);  //  0.7 MB
  bf16*  Wdt_b  = (bf16*) alloc((size_t)INNER_* R_     * 2);  //  0.5 MB
  bf16*  Wout_b = (bf16*) alloc((size_t)E_    * INNER_ * 2);  //  4.2 MB
  bf16*  xz_b   = (bf16*) alloc((size_t)BT_   * F_IN   * 2);  // 33.5 MB
  bf16*  u_b    = (bf16*) alloc((size_t)BT_   * INNER_ * 2);  // 16.8 MB
  float* dbl_f  = (float*)alloc((size_t)BT_   * F_X    * 4);  //  2.7 MB
  bf16*  dblR_b = (bf16*) alloc((size_t)BT_   * R_     * 2);  //  1.0 MB
  float* dt_f   = (float*)alloc((size_t)BT_   * INNER_ * 4);  // 33.5 MB
  bf16*  y_b    = (bf16*) alloc((size_t)BT_   * INNER_ * 2);  // 16.8 MB

  // Weight precision conversion (f32 -> bf16)
  cvt_bf16<<<(F_IN * E_ + 255) / 256,      256, 0, stream>>>(W_in,  Win_b,  F_IN * E_);
  cvt_bf16<<<(F_X * INNER_ + 255) / 256,   256, 0, stream>>>(Wx,    Wx_b,   F_X * INNER_);
  cvt_bf16<<<(INNER_ * R_ + 255) / 256,    256, 0, stream>>>(Wdt,   Wdt_b,  INNER_ * R_);
  cvt_bf16<<<(E_ * INNER_ + 255) / 256,    256, 0, stream>>>(W_out, Wout_b, E_ * INNER_);

  // 1) LayerNorm -> xn (bf16)
  ln_kernel<<<BT_, 256, 0, stream>>>(x, norm_g, norm_b, xn_b);

  // 2) xz = xn @ W_in^T  (4096x1024x4096), store bf16
  gemm_wmma<4, EPI_BF16><<<dim3(F_IN / 64, BT_ / 128), 256, 0, stream>>>(
      xn_b, Win_b, E_, nullptr, xz_b, nullptr, 0, F_IN);

  // 3) depthwise causal conv (K=4) + SiLU -> u (bf16)
  conv_silu_kernel<<<(BT_ * INNER_) / 256, 256, 0, stream>>>(xz_b, conv_w, conv_b, u_b);

  // 4) dbl = u @ Wx^T (4096x2048x160), f32 + bf16 copy of first R cols
  gemm_wmma<2, EPI_DBL><<<dim3(F_X / 32, BT_ / 128), 256, 0, stream>>>(
      u_b, Wx_b, INNER_, dbl_f, dblR_b, nullptr, F_X, R_);

  // 5) dt = softplus(dblR @ Wdt^T + bdt) (4096x128x2048), f32
  gemm_wmma<4, EPI_DT><<<dim3(INNER_ / 64, BT_ / 128), 256, 0, stream>>>(
      dblR_b, Wdt_b, R_, dt_f, nullptr, bdt, INNER_, 0);

  // 6) selective scan + D skip + silu(z) gate -> y (bf16)
  scan_kernel<<<(B_ * (INNER_ / 2) * 32) / 256, 256, 0, stream>>>(
      dt_f, u_b, dbl_f, xz_b, A_log, Dp, y_b);

  // 7) out = y @ W_out^T + residual (4096x2048x1024), f32
  gemm_wmma<4, EPI_RESID><<<dim3(E_ / 64, BT_ / 128), 256, 0, stream>>>(
      y_b, Wout_b, INNER_, out, nullptr, x, E_, 0);
}
